// TreeLSTM_15710990369455
// MI455X (gfx1250) — matrix operand order repaired and checked by reading, not compiled, not run on hardware
//
#include <hip/hip_runtime.h>

// ---------------------------------------------------------------- constants
constexpr int NB    = 32;                  // trees
constexpr int DEPTH = 13;
constexpr int NPT   = (1 << DEPTH) - 1;    // 8191 nodes / tree
constexpr int NTOT  = NB * NPT;            // 262112
constexpr int IND   = 128;                 // input dim
constexpr int HD    = 256;                 // hidden dim
constexpr int OUTD  = 256;

// ---------------------------------------------------------------- WMMA types
typedef __attribute__((ext_vector_type(16))) __bf16 v16bf;
typedef __attribute__((ext_vector_type(8)))  float  v8f;

union BFr { v16bf v; unsigned int u[8]; };

__device__ __forceinline__ unsigned short f2bf(float f) {
  unsigned int u = __float_as_uint(f);
  u += 0x7fffu + ((u >> 16) & 1u);            // round-to-nearest-even
  return (unsigned short)(u >> 16);
}
__device__ __forceinline__ unsigned int pk2(float a, float b) {
  return (unsigned int)f2bf(a) | ((unsigned int)f2bf(b) << 16);
}
// packed 2x bf16 add (CDNA5 VOP3P op 35)
__device__ __forceinline__ unsigned int pkadd_bf16(unsigned int a, unsigned int b) {
  unsigned int d;
  asm("v_pk_add_bf16 %0, %1, %2" : "=v"(d) : "v"(a), "v"(b));
  return d;
}
__device__ __forceinline__ float sigf(float x) { return 1.0f / (1.0f + __expf(-x)); }

__device__ __forceinline__ v8f wmma_bf16(const BFr& a, const BFr& b, v8f c) {
  return __builtin_amdgcn_wmma_f32_16x16x32_bf16(
      /*neg_a=*/false, a.v, /*neg_b=*/false, b.v,
      /*c_mod=*/(short)0, c, /*reuse_a=*/false, /*reuse_b=*/false);
}

// A fragment straight from a bf16 row in global memory.
// A k order used here (contiguous, not AMD's table order): lane<16 -> K 0..15; lane>=16 -> K 16..31,
// i.e. pair indices kb..kb+7 with kb = (lane&16)?8:0.
__device__ __forceinline__ void load_afrag_bf(BFr& a, const unsigned short* row,
                                              int k0, int lane) {
  const unsigned int* p =
      (const unsigned int*)row + (k0 >> 1) + ((lane & 16) ? 8 : 0);
  const uint4 lo = ((const uint4*)p)[0];
  const uint4 hi = ((const uint4*)(p + 4))[0];
  a.u[0] = lo.x; a.u[1] = lo.y; a.u[2] = lo.z; a.u[3] = lo.w;
  a.u[4] = hi.x; a.u[5] = hi.y; a.u[6] = hi.z; a.u[7] = hi.w;
}

// B fragment from packed transposed weight Wt[N][K] (bf16, pair stride = K/2).
// B k order used here (contiguous, as A): lane<16 -> K 0..15 ; lane>=16 -> K 16..31 (col = lane&15).
__device__ __forceinline__ void load_bfrag(BFr& b, const unsigned int* W,
                                           int coln, int kp0, int stridep) {
#pragma unroll
  for (int v = 0; v < 8; ++v) b.u[v] = W[(size_t)coln * stridep + kp0 + v];
}

// pack 16 consecutive f32 (4x float4) into 8 bf16-pairs
__device__ __forceinline__ void pack8f(const float4& a, const float4& b,
                                       const float4& c, const float4& d,
                                       unsigned int* o) {
  o[0] = pk2(a.x, a.y); o[1] = pk2(a.z, a.w);
  o[2] = pk2(b.x, b.y); o[3] = pk2(b.z, b.w);
  o[4] = pk2(c.x, c.y); o[5] = pk2(c.z, c.w);
  o[6] = pk2(d.x, d.y); o[7] = pk2(d.z, d.w);
}

// ---------------------------------------------------------------- weight pack
// dst[n*K + k] = bf16(src[k*NC + n])   (transpose + convert)
__global__ void pack_kernel(const float* __restrict__ src,
                            unsigned short* __restrict__ dst, int K, int NC) {
  int i = blockIdx.x * 256 + threadIdx.x;
  if (i >= K * NC) return;
  int n = i / K, k = i % K;
  dst[n * K + k] = f2bf(src[k * NC + n]);
}

// ---------------------------------------------------------------- input proj
// hb = bf16(relu(x @ Wp + bp))   x:[N,128] f32, WpT packed bf16 [256][128]
__global__ __launch_bounds__(128) void proj_kernel(
    const float* __restrict__ x, const unsigned int* __restrict__ WpT,
    const float* __restrict__ bp, unsigned short* __restrict__ hb) {
  const int tid = threadIdx.x, wave = tid >> 5, lane = tid & 31;
  const int row0 = blockIdx.x * 64, colblk = blockIdx.y * 64;

  // A-row for this lane (clamped for the partial last tile)
  const int arr = row0 + 16 * wave + (lane & 15);
  const int arow = (arr < NTOT) ? arr : NTOT - 1;
  const float* xrow = x + (size_t)arow * IND;

  v8f zero = {0, 0, 0, 0, 0, 0, 0, 0};
  v8f acc[4];
#pragma unroll
  for (int nt = 0; nt < 4; ++nt) acc[nt] = zero;

  for (int kt = 0; kt < 4; ++kt) {           // K = 128
    const int k0 = kt * 32;
    BFr af;
    {
      const float* p = xrow + k0 + ((lane & 16) ? 16 : 0);
      const float4 f0 = ((const float4*)p)[0];
      const float4 f1 = ((const float4*)p)[1];
      const float4 f2 = ((const float4*)(p + 8))[0];
      const float4 f3 = ((const float4*)(p + 8))[1];
      pack8f(f0, f1, f2, f3, af.u);
    }
    const int kp0 = (k0 >> 1) + ((lane & 16) ? 8 : 0);
#pragma unroll
    for (int nt = 0; nt < 4; ++nt) {
      BFr bw;
      load_bfrag(bw, WpT, colblk + nt * 16 + (lane & 15), kp0, 64);
      acc[nt] = wmma_bf16(af, bw, acc[nt]);
    }
  }
  const int rbase = row0 + 16 * wave + ((lane & 16) ? 8 : 0);
  if (row0 + 64 <= NTOT) {                 // fast path: full tile
#pragma unroll
    for (int nt = 0; nt < 4; ++nt) {
      const int coln = colblk + nt * 16 + (lane & 15);
      const float b = bp[coln];
#pragma unroll
      for (int v = 0; v < 8; ++v)
        hb[(size_t)(rbase + v) * HD + coln] = f2bf(fmaxf(acc[nt][v] + b, 0.0f));
    }
  } else {                                 // guarded tail tile
#pragma unroll
    for (int nt = 0; nt < 4; ++nt) {
      const int coln = colblk + nt * 16 + (lane & 15);
      const float b = bp[coln];
#pragma unroll
      for (int v = 0; v < 8; ++v) {
        const int rr = rbase + v;
        if (rr < NTOT)
          hb[(size_t)rr * HD + coln] = f2bf(fmaxf(acc[nt][v] + b, 0.0f));
      }
    }
  }
}

// ---------------------------------------------------------------- leaf level
// i=sig(xv@Wi+bi) o=sig(xv@Wo+bo) u=tanh(xv@Wu+bu); c=i*u; h=o*tanh(c)
__global__ __launch_bounds__(128) void tree_leaf_kernel(
    const unsigned short* __restrict__ hin, unsigned short* __restrict__ hnew,
    float* __restrict__ cbuf, float* __restrict__ fout,
    const unsigned int* __restrict__ Wi, const unsigned int* __restrict__ Wo,
    const unsigned int* __restrict__ Wu,
    const float* __restrict__ bi, const float* __restrict__ bo,
    const float* __restrict__ bu, int d) {
  const int tid = threadIdx.x, wave = tid >> 5, lane = tid & 31;
  const int P = 1 << d, M = NB << d;
  const int row0 = blockIdx.x * 64, colblk = blockIdx.y * 64;

  // A-row node index for this lane (clamped)
  const int arr0 = row0 + 16 * wave + (lane & 15);
  const int arr = (arr0 < M) ? arr0 : M - 1;
  const int aNd = (arr >> d) * NPT + (P - 1) + (arr & (P - 1));
  const unsigned short* xrow = hin + (size_t)aNd * HD;

  v8f zero = {0, 0, 0, 0, 0, 0, 0, 0};
  v8f ai[4], ao[4], au[4];
#pragma unroll
  for (int nt = 0; nt < 4; ++nt) { ai[nt] = zero; ao[nt] = zero; au[nt] = zero; }

  for (int kt = 0; kt < 8; ++kt) {           // K = 256
    const int k0 = kt * 32;
    BFr ax;
    load_afrag_bf(ax, xrow, k0, lane);
    const int kp0 = (k0 >> 1) + ((lane & 16) ? 8 : 0);
#pragma unroll
    for (int nt = 0; nt < 4; ++nt) {
      const int coln = colblk + nt * 16 + (lane & 15);
      BFr bw;
      load_bfrag(bw, Wi, coln, kp0, 128); ai[nt] = wmma_bf16(ax, bw, ai[nt]);
      load_bfrag(bw, Wo, coln, kp0, 128); ao[nt] = wmma_bf16(ax, bw, ao[nt]);
      load_bfrag(bw, Wu, coln, kp0, 128); au[nt] = wmma_bf16(ax, bw, au[nt]);
    }
  }

  // D-row node indices for this lane
  int nds[8];
  bool ok[8];
#pragma unroll
  for (int v = 0; v < 8; ++v) {
    const int rr = row0 + 16 * wave + ((lane & 16) ? v + 8 : v);
    const int rc = (rr < M) ? rr : M - 1;
    nds[v] = (rc >> d) * NPT + (P - 1) + (rc & (P - 1));
    ok[v] = (rr < M);
  }
  const bool full = (row0 + 64 <= M);
#pragma unroll
  for (int nt = 0; nt < 4; ++nt) {
    const int coln = colblk + nt * 16 + (lane & 15);
    const float Bi = bi[coln], Bo = bo[coln], Bu = bu[coln];
    if (full) {
#pragma unroll
      for (int v = 0; v < 8; ++v) {
        const size_t nd = (size_t)nds[v];
        const float iv = sigf(ai[nt][v] + Bi);
        const float ov = sigf(ao[nt][v] + Bo);
        const float uv = tanhf(au[nt][v] + Bu);
        const float cv = iv * uv;
        const float hv = ov * tanhf(cv);
        cbuf[nd * HD + coln] = cv;
        hnew[nd * HD + coln] = f2bf(hv);
        if (fout) fout[nd * HD + coln] = hv;
      }
    } else {
#pragma unroll
      for (int v = 0; v < 8; ++v) {
        if (!ok[v]) continue;
        const size_t nd = (size_t)nds[v];
        const float iv = sigf(ai[nt][v] + Bi);
        const float ov = sigf(ao[nt][v] + Bo);
        const float uv = tanhf(au[nt][v] + Bu);
        const float cv = iv * uv;
        const float hv = ov * tanhf(cv);
        cbuf[nd * HD + coln] = cv;
        hnew[nd * HD + coln] = f2bf(hv);
        if (fout) fout[nd * HD + coln] = hv;
      }
    }
  }
}

// ------------------------------------------------------------- internal level
// Fused: i=sig(xv@Wi+hs@Ui+bi), o,u analogous; fL=sig(xv@Wf+bf+hL@Uf),
// fR=sig(xv@Wf+bf+hR@Uf); c=i*u+fL*cL+fR*cR; h=o*tanh(c)
__global__ __launch_bounds__(128) void tree_node_kernel(
    const unsigned short* __restrict__ hin, unsigned short* __restrict__ hnew,
    float* __restrict__ cbuf, float* __restrict__ fout,
    const unsigned int* __restrict__ Wi, const unsigned int* __restrict__ Ui,
    const unsigned int* __restrict__ Wo, const unsigned int* __restrict__ Uo,
    const unsigned int* __restrict__ Wu, const unsigned int* __restrict__ Uu,
    const unsigned int* __restrict__ Wf, const unsigned int* __restrict__ Uf,
    const float* __restrict__ bi, const float* __restrict__ bo,
    const float* __restrict__ bu, const float* __restrict__ bfv, int d) {
  const int tid = threadIdx.x, wave = tid >> 5, lane = tid & 31;
  const int P = 1 << d, M = NB << d;
  const int row0 = blockIdx.x * 64, colblk = blockIdx.y * 64;

  // A-row node/child indices for this lane (clamped)
  const int arr0 = row0 + 16 * wave + (lane & 15);
  const int arr = (arr0 < M) ? arr0 : M - 1;
  const int am = (P - 1) + (arr & (P - 1));
  const int ab = (arr >> d) * NPT;
  const unsigned short* xrow = hin  + (size_t)(ab + am) * HD;
  const unsigned short* lrow = hnew + (size_t)(ab + 2 * am + 1) * HD;
  const unsigned short* rrow = hnew + (size_t)(ab + 2 * am + 2) * HD;

  v8f zero = {0, 0, 0, 0, 0, 0, 0, 0};
  v8f ai[4], ao[4], au[4], af[4], afl[4], afr[4];
#pragma unroll
  for (int nt = 0; nt < 4; ++nt) {
    ai[nt] = zero; ao[nt] = zero; au[nt] = zero;
    af[nt] = zero; afl[nt] = zero; afr[nt] = zero;
  }

  for (int kt = 0; kt < 8; ++kt) {           // K = 256
    const int k0 = kt * 32;
    BFr ax, hl, hr, hs;
    load_afrag_bf(ax, xrow, k0, lane);
    load_afrag_bf(hl, lrow, k0, lane);
    load_afrag_bf(hr, rrow, k0, lane);
#pragma unroll
    for (int v = 0; v < 8; ++v) hs.u[v] = pkadd_bf16(hl.u[v], hr.u[v]);
    const int kp0 = (k0 >> 1) + ((lane & 16) ? 8 : 0);
#pragma unroll
    for (int nt = 0; nt < 4; ++nt) {
      const int coln = colblk + nt * 16 + (lane & 15);
      BFr bw;
      load_bfrag(bw, Wi, coln, kp0, 128); ai[nt]  = wmma_bf16(ax, bw, ai[nt]);
      load_bfrag(bw, Ui, coln, kp0, 128); ai[nt]  = wmma_bf16(hs, bw, ai[nt]);
      load_bfrag(bw, Wo, coln, kp0, 128); ao[nt]  = wmma_bf16(ax, bw, ao[nt]);
      load_bfrag(bw, Uo, coln, kp0, 128); ao[nt]  = wmma_bf16(hs, bw, ao[nt]);
      load_bfrag(bw, Wu, coln, kp0, 128); au[nt]  = wmma_bf16(ax, bw, au[nt]);
      load_bfrag(bw, Uu, coln, kp0, 128); au[nt]  = wmma_bf16(hs, bw, au[nt]);
      load_bfrag(bw, Wf, coln, kp0, 128); af[nt]  = wmma_bf16(ax, bw, af[nt]);
      load_bfrag(bw, Uf, coln, kp0, 128);
      afl[nt] = wmma_bf16(hl, bw, afl[nt]);
      afr[nt] = wmma_bf16(hr, bw, afr[nt]);
    }
  }

  // D-row node/child indices for this lane (clamped)
  int nds[8], cls[8];
  bool ok[8];
#pragma unroll
  for (int v = 0; v < 8; ++v) {
    const int rr = row0 + 16 * wave + ((lane & 16) ? v + 8 : v);
    const int rc = (rr < M) ? rr : M - 1;
    const int m = (P - 1) + (rc & (P - 1));
    const int b = (rc >> d) * NPT;
    nds[v] = b + m;
    cls[v] = b + 2 * m + 1;
    ok[v] = (rr < M);
  }
  const bool full = (row0 + 64 <= M);
#pragma unroll
  for (int nt = 0; nt < 4; ++nt) {
    const int coln = colblk + nt * 16 + (lane & 15);
    const float Bi = bi[coln], Bo = bo[coln], Bu = bu[coln], Bf = bfv[coln];
    if (full) {
#pragma unroll
      for (int v = 0; v < 8; ++v) {
        const size_t nd = (size_t)nds[v];
        const size_t cl = (size_t)cls[v];
        const float iv = sigf(ai[nt][v] + Bi);
        const float ov = sigf(ao[nt][v] + Bo);
        const float uv = tanhf(au[nt][v] + Bu);
        const float fpre = af[nt][v] + Bf;
        const float fl = sigf(fpre + afl[nt][v]);
        const float fr = sigf(fpre + afr[nt][v]);
        const float cL = cbuf[cl * HD + coln];
        const float cR = cbuf[(cl + 1) * HD + coln];
        const float cv = iv * uv + fl * cL + fr * cR;
        const float hv = ov * tanhf(cv);
        cbuf[nd * HD + coln] = cv;
        hnew[nd * HD + coln] = f2bf(hv);
        if (fout) fout[nd * HD + coln] = hv;
      }
    } else {
#pragma unroll
      for (int v = 0; v < 8; ++v) {
        if (!ok[v]) continue;
        const size_t nd = (size_t)nds[v];
        const size_t cl = (size_t)cls[v];
        const float iv = sigf(ai[nt][v] + Bi);
        const float ov = sigf(ao[nt][v] + Bo);
        const float uv = tanhf(au[nt][v] + Bu);
        const float fpre = af[nt][v] + Bf;
        const float fl = sigf(fpre + afl[nt][v]);
        const float fr = sigf(fpre + afr[nt][v]);
        const float cL = cbuf[cl * HD + coln];
        const float cR = cbuf[(cl + 1) * HD + coln];
        const float cv = iv * uv + fl * cL + fr * cR;
        const float hv = ov * tanhf(cv);
        cbuf[nd * HD + coln] = cv;
        hnew[nd * HD + coln] = f2bf(hv);
        if (fout) fout[nd * HD + coln] = hv;
      }
    }
  }
}

// ---------------------------------------------------------------- attention
__global__ void attn_logits_kernel(const float* __restrict__ h,
                                   const float* __restrict__ Wa,
                                   const float* __restrict__ ba,
                                   float* __restrict__ logits) {
  const int lane = threadIdx.x & 31;
  const int wid = (blockIdx.x * blockDim.x + threadIdx.x) >> 5;
  const int nw = (gridDim.x * blockDim.x) >> 5;
  for (int n = wid; n < NTOT; n += nw) {
    float s = 0.0f;
#pragma unroll
    for (int k = lane; k < HD; k += 32) s += h[(size_t)n * HD + k] * Wa[k];
    for (int o = 16; o > 0; o >>= 1) s += __shfl_down(s, o, 32);
    if (lane == 0) logits[n] = s + ba[0];
  }
}

__global__ void reduce_max_kernel(const float* __restrict__ x, int n,
                                  float* __restrict__ part) {
  __shared__ float sm[256];
  float m = -3.4e38f;
  for (int i = blockIdx.x * blockDim.x + threadIdx.x; i < n;
       i += gridDim.x * blockDim.x)
    m = fmaxf(m, x[i]);
  sm[threadIdx.x] = m;
  __syncthreads();
  for (int s = 128; s > 0; s >>= 1) {
    if (threadIdx.x < (unsigned)s)
      sm[threadIdx.x] = fmaxf(sm[threadIdx.x], sm[threadIdx.x + s]);
    __syncthreads();
  }
  if (threadIdx.x == 0) part[blockIdx.x] = sm[0];
}

__global__ void final_max_kernel(const float* __restrict__ part,
                                 float* __restrict__ scal) {
  __shared__ float sm[256];
  sm[threadIdx.x] = part[threadIdx.x];
  __syncthreads();
  for (int s = 128; s > 0; s >>= 1) {
    if (threadIdx.x < (unsigned)s)
      sm[threadIdx.x] = fmaxf(sm[threadIdx.x], sm[threadIdx.x + s]);
    __syncthreads();
  }
  if (threadIdx.x == 0) scal[0] = sm[0];
}

__global__ void reduce_sum_kernel(const float* __restrict__ x, int n,
                                  const float* __restrict__ scal,
                                  float* __restrict__ part) {
  __shared__ float sm[256];
  const float gm = scal[0];
  float s = 0.0f;
  for (int i = blockIdx.x * blockDim.x + threadIdx.x; i < n;
       i += gridDim.x * blockDim.x)
    s += __expf(x[i] - gm);
  sm[threadIdx.x] = s;
  __syncthreads();
  for (int t = 128; t > 0; t >>= 1) {
    if (threadIdx.x < (unsigned)t) sm[threadIdx.x] += sm[threadIdx.x + t];
    __syncthreads();
  }
  if (threadIdx.x == 0) part[blockIdx.x] = sm[0];
}

__global__ void final_sum_kernel(const float* __restrict__ part,
                                 float* __restrict__ scal) {
  __shared__ float sm[256];
  sm[threadIdx.x] = part[threadIdx.x];
  __syncthreads();
  for (int t = 128; t > 0; t >>= 1) {
    if (threadIdx.x < (unsigned)t) sm[threadIdx.x] += sm[threadIdx.x + t];
    __syncthreads();
  }
  if (threadIdx.x == 0) scal[1] = sm[0];
}

__global__ void aw_kernel(const float* __restrict__ logits,
                          const float* __restrict__ scal,
                          float* __restrict__ aw) {
  const int i = blockIdx.x * 256 + threadIdx.x;
  if (i < NTOT) aw[i] = __expf(logits[i] - scal[0]) * (1.0f / scal[1]);
}

// chunked deterministic segment-sum pooling (no atomics)
__global__ void pool_kernel(const float* __restrict__ h,
                            const float* __restrict__ aw,
                            float* __restrict__ part) {
  const int b = blockIdx.x, chunk = blockIdx.y, col = threadIdx.x;
  const int start = chunk * 1024;
  const int end = (start + 1024 < NPT) ? start + 1024 : NPT;
  float acc = 0.0f;
  for (int m = start; m < end; ++m) {
    const int n = b * NPT + m;
    acc += h[(size_t)n * HD + col] * aw[n];
  }
  part[((size_t)b * 8 + chunk) * HD + col] = acc;
}

__global__ void pool_reduce_kernel(const float* __restrict__ part,
                                   float* __restrict__ pooled) {
  const int b = blockIdx.x, col = threadIdx.x;
  float s = 0.0f;
#pragma unroll
  for (int c = 0; c < 8; ++c) s += part[((size_t)b * 8 + c) * HD + col];
  pooled[b * HD + col] = s;
}

__global__ void out_kernel(const float* __restrict__ pooled,
                           const float* __restrict__ Wout,
                           const float* __restrict__ bout,
                           float* __restrict__ out) {
  const int b = blockIdx.x, j = threadIdx.x;
  float s = bout[j];
  for (int k = 0; k < HD; ++k) s += pooled[b * HD + k] * Wout[k * OUTD + j];
  out[b * OUTD + j] = s;
}

// ---------------------------------------------------------------- launcher
extern "C" void kernel_launch(void* const* d_in, const int* in_sizes, int n_in,
                              void* d_out, int out_size, void* d_ws,
                              size_t ws_size, hipStream_t stream) {
  (void)in_sizes; (void)n_in; (void)out_size; (void)ws_size;

  const float* x    = (const float*)d_in[0];
  const float* Wp   = (const float*)d_in[1];
  const float* bp   = (const float*)d_in[2];
  const float* Wa   = (const float*)d_in[15];
  const float* ba   = (const float*)d_in[16];
  const float* Wout = (const float*)d_in[17];
  const float* bout = (const float*)d_in[18];

  float* out     = (float*)d_out;
  float* d_nodes = out + NB * OUTD;                  // node_embeddings region
  float* d_attn  = d_nodes + (size_t)NTOT * HD;      // attention_weights region

  char* base = (char*)d_ws;
  size_t off = 0;
  auto carve = [&](size_t bytes) -> void* {
    void* p = base + off;
    off = (off + bytes + 255) & ~(size_t)255;
    return p;
  };
  unsigned short* hb0 =
      (unsigned short*)carve((size_t)NTOT * HD * sizeof(unsigned short));
  unsigned short* hb1 =
      (unsigned short*)carve((size_t)NTOT * HD * sizeof(unsigned short));
  float* cbuf   = (float*)carve((size_t)NTOT * HD * sizeof(float));
  float* logits = (float*)carve((size_t)NTOT * sizeof(float));
  float* part   = (float*)carve(256 * sizeof(float));
  float* scal   = (float*)carve(16);
  float* ppart  = (float*)carve((size_t)NB * 8 * HD * sizeof(float));
  float* pooled = (float*)carve((size_t)NB * HD * sizeof(float));
  unsigned short* packs =
      (unsigned short*)carve((size_t)(IND * HD + 16 * HD * HD) * sizeof(unsigned short));

  // packed transposed bf16 weights: Wp, then [layer][Wi,Ui,Wo,Uo,Wu,Uu,Wf,Uf]
  unsigned short* wpT = packs;
  unsigned short* mt[2][8];
  {
    size_t o = (size_t)IND * HD;
    for (int l = 0; l < 2; ++l)
      for (int m = 0; m < 8; ++m) { mt[l][m] = packs + o; o += (size_t)HD * HD; }
  }

  pack_kernel<<<(IND * HD + 255) / 256, 256, 0, stream>>>(Wp, wpT, IND, HD);
  const int widx[8] = {3, 5, 6, 8, 9, 11, 12, 14};   // Wi Ui Wo Uo Wu Uu Wf Uf
  for (int l = 0; l < 2; ++l)
    for (int m = 0; m < 8; ++m)
      pack_kernel<<<(HD * HD + 255) / 256, 256, 0, stream>>>(
          (const float*)d_in[widx[m]] + (size_t)l * HD * HD, mt[l][m], HD, HD);

  // input projection -> hb0 (layer-0 input, bf16)
  {
    dim3 g((NTOT + 63) / 64, 4);
    proj_kernel<<<g, 128, 0, stream>>>(x, (const unsigned int*)wpT, bp, hb0);
  }

  for (int l = 0; l < 2; ++l) {
    const unsigned short* hin = (l == 0) ? hb0 : hb1;
    unsigned short* hout      = (l == 0) ? hb1 : hb0;
    float* fout               = (l == 0) ? nullptr : d_nodes;  // final f32 h
    const float* bi = (const float*)d_in[4]  + (size_t)l * HD;
    const float* bo = (const float*)d_in[7]  + (size_t)l * HD;
    const float* bu = (const float*)d_in[10] + (size_t)l * HD;
    const float* bf = (const float*)d_in[13] + (size_t)l * HD;
    {
      const int dd = DEPTH - 1, M = NB << dd;
      dim3 g((M + 63) / 64, 4);
      tree_leaf_kernel<<<g, 128, 0, stream>>>(
          hin, hout, cbuf, fout,
          (const unsigned int*)mt[l][0], (const unsigned int*)mt[l][2],
          (const unsigned int*)mt[l][4], bi, bo, bu, dd);
    }
    for (int dd = DEPTH - 2; dd >= 0; --dd) {
      const int M = NB << dd;
      dim3 g((M + 63) / 64, 4);
      tree_node_kernel<<<g, 128, 0, stream>>>(
          hin, hout, cbuf, fout,
          (const unsigned int*)mt[l][0], (const unsigned int*)mt[l][1],
          (const unsigned int*)mt[l][2], (const unsigned int*)mt[l][3],
          (const unsigned int*)mt[l][4], (const unsigned int*)mt[l][5],
          (const unsigned int*)mt[l][6], (const unsigned int*)mt[l][7],
          bi, bo, bu, bf, dd);
    }
  }

  attn_logits_kernel<<<1024, 256, 0, stream>>>(d_nodes, Wa, ba, logits);
  reduce_max_kernel<<<256, 256, 0, stream>>>(logits, NTOT, part);
  final_max_kernel<<<1, 256, 0, stream>>>(part, scal);
  reduce_sum_kernel<<<256, 256, 0, stream>>>(logits, NTOT, scal, part);
  final_sum_kernel<<<1, 256, 0, stream>>>(part, scal);
  aw_kernel<<<(NTOT + 255) / 256, 256, 0, stream>>>(logits, scal, d_attn);
  pool_kernel<<<dim3(NB, 8), 256, 0, stream>>>(d_nodes, d_attn, ppart);
  pool_reduce_kernel<<<NB, 256, 0, stream>>>(ppart, pooled);
  out_kernel<<<NB, 256, 0, stream>>>(pooled, Wout, bout, out);
}
